// Attention_90855738179837
// MI455X (gfx1250) — matrix operand order repaired and checked by reading, not compiled, not run on hardware
//
#include <hip/hip_runtime.h>
#include <hip/hip_bf16.h>
#include <stdint.h>

// ---------------------------------------------------------------------------
// Attention (B=4, S=2048, E=Dk=Dv=1024), softmax over QUERY axis.
// All matmuls: bf16 WMMA (v_wmma_f32_16x16x32_bf16) with f32 accumulation.
// Column softmax (axis=q) folded into V via per-column reciprocal.
// Compute-bound (~120 GFLOP vs ~100-200MB HBM traffic @ 23.3 TB/s).
// Staging: gfx1250 ASYNC-to-LDS DMA (GLOBAL_LOAD_ASYNC_TO_LDS_B128 +
// s_wait_asynccnt) -- builtin confirmed present; takes (AS1 v4i*, AS3 v4i*,
// imm offset, imm cpol). Register-pipelined fallback kept under #if.
// ---------------------------------------------------------------------------

#define B_   4
#define S_   2048
#define E_   1024
#define DK_  1024
#define DV_  1024

typedef __attribute__((ext_vector_type(16))) __bf16          v16bf;
typedef __attribute__((ext_vector_type(8)))  float           v8f;
typedef __attribute__((ext_vector_type(4)))  unsigned int    v4u;
typedef __attribute__((ext_vector_type(4)))  int             v4i;
typedef __attribute__((ext_vector_type(4)))  float           v4f;
typedef __attribute__((ext_vector_type(4)))  unsigned short  v4us;

union FragBF { v4u u[2]; v16bf v; };

#if __has_builtin(__builtin_amdgcn_global_load_async_to_lds_b128)
#define USE_ASYNC_LDS 1
typedef __attribute__((address_space(1))) v4i gv4i;   // global 16B chunk
typedef __attribute__((address_space(3))) v4i lv4i;   // LDS 16B chunk
#else
#define USE_ASYNC_LDS 0
#endif

__device__ __forceinline__ void wait_async_zero() {
#if USE_ASYNC_LDS
#if __has_builtin(__builtin_amdgcn_s_wait_asynccnt)
  __builtin_amdgcn_s_wait_asynccnt(0);
#else
  asm volatile("s_wait_asynccnt 0x0" ::: "memory");
#endif
#endif
}

__device__ __forceinline__ unsigned short f32_to_bf16(float f) {
  unsigned int u = __float_as_uint(f);
  u += 0x7FFFu + ((u >> 16) & 1u);            // round-to-nearest-even
  return (unsigned short)(u >> 16);
}
__device__ __forceinline__ float bf16_to_f32(unsigned short h) {
  return __uint_as_float(((unsigned int)h) << 16);
}

// ---------------------------------------------------------------------------
// Elementwise fp32 -> bf16 (vectorized x4)
// ---------------------------------------------------------------------------
__global__ __launch_bounds__(256) void cvt_f32_to_bf16(
    const float* __restrict__ in, unsigned short* __restrict__ out, int n4) {
  int i = blockIdx.x * 256 + threadIdx.x;
  if (i >= n4) return;
  v4f f = reinterpret_cast<const v4f*>(in)[i];
  v4us o;
  o[0] = f32_to_bf16(f[0]);
  o[1] = f32_to_bf16(f[1]);
  o[2] = f32_to_bf16(f[2]);
  o[3] = f32_to_bf16(f[3]);
  reinterpret_cast<v4us*>(out)[i] = o;
}

// ---------------------------------------------------------------------------
// Transpose + convert: W fp32 [K][N] -> Wt bf16 [N][K]
// ---------------------------------------------------------------------------
__global__ __launch_bounds__(256) void transpose_w_f32_to_bf16(
    const float* __restrict__ W, unsigned short* __restrict__ Wt, int K, int N) {
  __shared__ float t[32][33];
  int n0 = blockIdx.x * 32, k0 = blockIdx.y * 32;
  int tx = threadIdx.x, ty = threadIdx.y;
  #pragma unroll
  for (int s = 0; s < 4; ++s) {
    int i = ty + s * 8;
    t[i][tx] = W[(long long)(k0 + i) * N + n0 + tx];
  }
  __syncthreads();
  #pragma unroll
  for (int s = 0; s < 4; ++s) {
    int i = ty + s * 8;
    Wt[(long long)(n0 + i) * K + k0 + tx] = f32_to_bf16(t[tx][i]);
  }
}

// ---------------------------------------------------------------------------
// bf16 WMMA GEMM:  C[z] = A[z] (M x K, row major) * Bt[z]^T (Bt is N x K)
//                         [* scale] [+ bias], C is f32 or bf16.
// Block = 256 threads = 8 wave32s, 128x128 C tile, K-step 32.
// Wave grid 4(M) x 2(N): each wave 32x64 = 2x4 frags = 8 WMMA / K-step.
// LDS double buffered; row stride padded to 80B (5 x uint4).
// Staging: async-to-LDS DMA (ASYNCcnt) or register-pipelined fallback.
// ---------------------------------------------------------------------------
template <bool OUT_BF16, bool HAS_BIAS>
__global__ __launch_bounds__(256) void gemm_wmma_bf16(
    const unsigned short* __restrict__ A,
    const unsigned short* __restrict__ Bt,
    const float* __restrict__ bias,
    void* __restrict__ C,
    int M, int N, int K,
    int lda, int ldb, int ldc,
    long long sA, long long sB, long long sC,
    float scale) {
  (void)M; (void)N;
  __shared__ v4u ldsA[2][128 * 5];
  __shared__ v4u ldsB[2][128 * 5];

  const int tid   = threadIdx.x;
  const int lane  = tid & 31;
  const int lh    = lane & 15;   // lane low half index
  const int hs    = lane >> 4;   // 0: lanes 0-15, 1: lanes 16-31
  const int wid   = tid >> 5;
  const int waveM = wid & 3;     // 32-row slab
  const int waveN = wid >> 2;    // 64-col slab

  const int mBase = blockIdx.y * 128;
  const int nBase = blockIdx.x * 128;
  const int z     = blockIdx.z;

  const unsigned short* Ab  = A  + (long long)z * sA;
  const unsigned short* Btb = Bt + (long long)z * sB;

  v8f acc[2][4];
  const v8f zero8 = {0.f, 0.f, 0.f, 0.f, 0.f, 0.f, 0.f, 0.f};
  #pragma unroll
  for (int mt = 0; mt < 2; ++mt)
    #pragma unroll
    for (int nt = 0; nt < 4; ++nt) acc[mt][nt] = zero8;

  auto compute = [&](int buf) {
    FragBF a[2], b[4];
    #pragma unroll
    for (int mt = 0; mt < 2; ++mt) {
      int r = waveM * 32 + mt * 16 + lh;            // A row in tile
      a[mt].u[0] = ldsA[buf][r * 5 + hs];           // K 0..7  / 8..15
      a[mt].u[1] = ldsA[buf][r * 5 + hs + 2];       // K 16..23 / 24..31
    }
    #pragma unroll
    for (int nt = 0; nt < 4; ++nt) {
      int col = waveN * 64 + nt * 16 + lh;          // B column in tile
      b[nt].u[0] = ldsB[buf][col * 5 + hs];         // K 0..7  / 8..15
      b[nt].u[1] = ldsB[buf][col * 5 + hs + 2];     // K 16..23 / 24..31
    }
    #pragma unroll
    for (int mt = 0; mt < 2; ++mt)
      #pragma unroll
      for (int nt = 0; nt < 4; ++nt)
        acc[mt][nt] = __builtin_amdgcn_wmma_f32_16x16x32_bf16(
            false, a[mt].v, false, b[nt].v, (short)0, acc[mt][nt], false, false);
  };

#if USE_ASYNC_LDS
  // -------- gfx1250 async DMA staging: no VGPR round-trip, ASYNCcnt-tracked
  auto stage = [&](int buf, int kk) {
    #pragma unroll
    for (int i = 0; i < 2; ++i) {
      int c   = tid * 2 + i;        // 0..511 : 128 rows x 4 16B-chunks
      int row = c >> 2;
      int cir = c & 3;
      __builtin_amdgcn_global_load_async_to_lds_b128(
          (gv4i*)(Ab + (long long)(mBase + row) * lda + kk + cir * 8),
          (lv4i*)&ldsA[buf][row * 5 + cir], 0, 0);
      __builtin_amdgcn_global_load_async_to_lds_b128(
          (gv4i*)(Btb + (long long)(nBase + row) * ldb + kk + cir * 8),
          (lv4i*)&ldsB[buf][row * 5 + cir], 0, 0);
    }
  };

  stage(0, 0);
  wait_async_zero();
  __syncthreads();
  for (int kk = 0; kk < K; kk += 32) {
    int cur = (kk >> 5) & 1;
    if (kk + 32 < K) stage(cur ^ 1, kk + 32);   // DMA overlaps with WMMA below
    compute(cur);
    wait_async_zero();                          // my DMAs into cur^1 landed
    __syncthreads();                            // everyone's landed / cur consumed
  }
#else
  // -------- register-pipelined staging: loads issued before compute,
  // ds_store (and its s_wait_loadcnt) only after the WMMAs.
  auto gload = [&](int kk, v4u r[4]) {
    #pragma unroll
    for (int i = 0; i < 2; ++i) {
      int c = tid * 2 + i, row = c >> 2, cir = c & 3;
      r[i]     = reinterpret_cast<const v4u*>(
                     Ab + (long long)(mBase + row) * lda + kk)[cir];
      r[2 + i] = reinterpret_cast<const v4u*>(
                     Btb + (long long)(nBase + row) * ldb + kk)[cir];
    }
  };
  auto lstore = [&](int buf, const v4u r[4]) {
    #pragma unroll
    for (int i = 0; i < 2; ++i) {
      int c = tid * 2 + i, row = c >> 2, cir = c & 3;
      ldsA[buf][row * 5 + cir] = r[i];
      ldsB[buf][row * 5 + cir] = r[2 + i];
    }
  };

  v4u rg[4];
  gload(0, rg);
  lstore(0, rg);
  __syncthreads();
  for (int kk = 0; kk < K; kk += 32) {
    int cur = (kk >> 5) & 1;
    bool more = (kk + 32 < K);
    if (more) gload(kk + 32, rg);   // issue global loads (no wait yet)
    compute(cur);                   // WMMAs hide the global latency
    if (more) lstore(cur ^ 1, rg);  // wait_loadcnt + ds_store after compute
    __syncthreads();
  }
#endif

  // Epilogue: 16x16 f32 C frag layout: VGPR r -> M = r + 8*hs, N = lane&15
  #pragma unroll
  for (int nt = 0; nt < 4; ++nt) {
    int col = nBase + waveN * 64 + nt * 16 + lh;
    float bval = 0.0f;
    if constexpr (HAS_BIAS) bval = bias[col];
    #pragma unroll
    for (int mt = 0; mt < 2; ++mt) {
      int rbase = mBase + waveM * 32 + mt * 16 + hs * 8;
      #pragma unroll
      for (int r = 0; r < 8; ++r) {
        float val = acc[mt][nt][r] * scale + bval;
        long long idx = (long long)z * sC + (long long)(rbase + r) * ldc + col;
        if constexpr (OUT_BF16)
          ((unsigned short*)C)[idx] = f32_to_bf16(val);
        else
          ((float*)C)[idx] = val;
      }
    }
  }
}

// ---------------------------------------------------------------------------
// Softmax over the QUERY axis: per (b,k) column of scores[b][q][k].
// Each thread owns one column -> no cross-lane reduction.
// Writes unnormalized exp(s - max) as bf16 P and 1/colsum.
// ---------------------------------------------------------------------------
__global__ __launch_bounds__(256) void softmax_axis_q(
    const float* __restrict__ scores, unsigned short* __restrict__ P,
    float* __restrict__ colRecip) {
  int b = blockIdx.y;
  int k = blockIdx.x * 256 + threadIdx.x;
  const float* sc     = scores + (long long)b * S_ * S_;
  unsigned short* p   = P      + (long long)b * S_ * S_;
  float m = -3.4e38f;
  for (int q = 0; q < S_; ++q) m = fmaxf(m, sc[(long long)q * S_ + k]);
  float sum = 0.f;
  for (int q = 0; q < S_; ++q) {
    float e = __expf(sc[(long long)q * S_ + k] - m);
    sum += e;
    p[(long long)q * S_ + k] = f32_to_bf16(e);
  }
  colRecip[b * S_ + k] = 1.0f / sum;
}

// ---------------------------------------------------------------------------
// Vt[b][v][k] = V[b][k][v] * colRecip[b][k]  (folds softmax denominator in)
// ---------------------------------------------------------------------------
__global__ __launch_bounds__(256) void transpose_scale_v(
    const unsigned short* __restrict__ V, const float* __restrict__ colRecip,
    unsigned short* __restrict__ Vt) {
  __shared__ float t[32][33];
  int b  = blockIdx.z;
  int v0 = blockIdx.x * 32, k0 = blockIdx.y * 32;
  int tx = threadIdx.x, ty = threadIdx.y;
  const unsigned short* Vb = V + (long long)b * S_ * DV_;
  unsigned short* Vtb      = Vt + (long long)b * DV_ * S_;
  #pragma unroll
  for (int s = 0; s < 4; ++s) {
    int i = ty + s * 8;
    float r = colRecip[b * S_ + k0 + i];
    t[i][tx] = bf16_to_f32(Vb[(long long)(k0 + i) * DV_ + v0 + tx]) * r;
  }
  __syncthreads();
  #pragma unroll
  for (int s = 0; s < 4; ++s) {
    int i = ty + s * 8;
    Vtb[(long long)(v0 + i) * S_ + k0 + tx] = f32_to_bf16(t[tx][i]);
  }
}

// ---------------------------------------------------------------------------
// Host-side orchestration
// ---------------------------------------------------------------------------
extern "C" void kernel_launch(void* const* d_in, const int* in_sizes, int n_in,
                              void* d_out, int out_size, void* d_ws, size_t ws_size,
                              hipStream_t stream) {
  (void)in_sizes; (void)n_in; (void)out_size; (void)ws_size;
  const float* x  = (const float*)d_in[0];
  const float* Wq = (const float*)d_in[1];
  const float* bq = (const float*)d_in[2];
  const float* Wk = (const float*)d_in[3];
  const float* bk = (const float*)d_in[4];
  const float* Wv = (const float*)d_in[5];
  const float* bv = (const float*)d_in[6];
  float* out = (float*)d_out;

  char* ws = (char*)d_ws;
  const size_t MB = 1ull << 20;
  // Workspace layout (aliased; peak ~134 MB):
  //   [0,16M)    x_bf   (bf16)  -> reused as Vt after projections
  //   [16,22M)   Wt q/k/v (bf16, [N][K])
  //   [22,54M)   Q,K    (bf16)  -> reused as P after scores
  //   [54,70M)   V      (bf16)
  //   [70,134M)  scores (f32)
  //   [134M,+32K) colRecip (f32)
  unsigned short* x_bf = (unsigned short*)(ws + 0);
  unsigned short* Wt0  = (unsigned short*)(ws + 16 * MB);
  unsigned short* Wt1  = (unsigned short*)(ws + 18 * MB);
  unsigned short* Wt2  = (unsigned short*)(ws + 20 * MB);
  unsigned short* Q    = (unsigned short*)(ws + 22 * MB);
  unsigned short* Kb   = (unsigned short*)(ws + 38 * MB);
  unsigned short* V    = (unsigned short*)(ws + 54 * MB);
  float*          Sc   = (float*)(ws + 70 * MB);
  unsigned short* P    = Q;      // aliases Q+K (32 MB), dead after scores
  unsigned short* Vt   = x_bf;   // aliases x_bf (16 MB), dead after projections
  float*          colRecip = (float*)(ws + 134 * MB);

  // 1) x -> bf16
  {
    int n4 = (B_ * S_ * E_) / 4;
    cvt_f32_to_bf16<<<n4 / 256, 256, 0, stream>>>(x, x_bf, n4);
  }
  // 2) W -> Wt bf16 [N][K]
  {
    dim3 blk(32, 8), grd(DK_ / 32, E_ / 32);
    transpose_w_f32_to_bf16<<<grd, blk, 0, stream>>>(Wq, Wt0, E_, DK_);
    transpose_w_f32_to_bf16<<<grd, blk, 0, stream>>>(Wk, Wt1, E_, DK_);
    transpose_w_f32_to_bf16<<<grd, blk, 0, stream>>>(Wv, Wt2, E_, DV_);
  }
  // 3) Q/K/V projections: [8192 x 1024] = x_bf[8192 x 1024] * W^T + bias
  {
    dim3 grd(DK_ / 128, (B_ * S_) / 128, 1);
    gemm_wmma_bf16<true, true><<<grd, 256, 0, stream>>>(
        x_bf, Wt0, bq, Q, B_ * S_, DK_, E_, E_, E_, DK_, 0, 0, 0, 1.0f);
    gemm_wmma_bf16<true, true><<<grd, 256, 0, stream>>>(
        x_bf, Wt1, bk, Kb, B_ * S_, DK_, E_, E_, E_, DK_, 0, 0, 0, 1.0f);
    gemm_wmma_bf16<true, true><<<grd, 256, 0, stream>>>(
        x_bf, Wt2, bv, V, B_ * S_, DV_, E_, E_, E_, DV_, 0, 0, 0, 1.0f);
  }
  // 4) scores[b] = (Q[b] * K[b]^T) / sqrt(Dk)   (K is already [N][Kdim])
  {
    dim3 grd(S_ / 128, S_ / 128, B_);
    gemm_wmma_bf16<false, false><<<grd, 256, 0, stream>>>(
        Q, Kb, nullptr, Sc, S_, S_, DK_, DK_, DK_, S_,
        (long long)S_ * DK_, (long long)S_ * DK_, (long long)S_ * S_, 0.03125f);
  }
  // 5) softmax over q (columns) -> unnormalized exp (bf16) + 1/colsum
  softmax_axis_q<<<dim3(S_ / 256, B_), 256, 0, stream>>>(Sc, P, colRecip);
  // 6) Vt[b][v][k] = V[b][k][v] / colsum[b][k]
  transpose_scale_v<<<dim3(DV_ / 32, S_ / 32, B_), dim3(32, 8), 0, stream>>>(
      V, colRecip, Vt);
  // 7) out[b] = P[b] * Vt[b]^T   (f32 output)
  {
    dim3 grd(DV_ / 128, S_ / 128, B_);
    gemm_wmma_bf16<false, false><<<grd, 256, 0, stream>>>(
        P, Vt, nullptr, out, S_, DV_, S_, S_, S_, DV_,
        (long long)S_ * S_, (long long)DV_ * S_, (long long)S_ * DV_, 1.0f);
  }
}
